// SparseMixtureOfExperts_14705968022236
// MI455X (gfx1250) — matrix operand order repaired and checked by reading, not compiled, not run on hardware
//
#include <hip/hip_runtime.h>
#include <hip/hip_bf16.h>
#include <stdint.h>

// ---------------------------------------------------------------------------
// Sparse MoE, MI455X (gfx1250, wave32, WMMA + TDM).
// N=32768 tokens, D=512, E=8 experts, K=2 top-k.
// gate fp32 -> bucket per expert -> grouped bf16 WMMA GEMM (f32 accum).
// B (weight) panel per block is DMA'd once into LDS by the Tensor Data Mover.
// ---------------------------------------------------------------------------

#define MOE_N 32768
#define MOE_D 512
#define MOE_E 8

typedef __attribute__((ext_vector_type(16))) __bf16 v16bf;
typedef __attribute__((ext_vector_type(8)))  float  v8f;
typedef unsigned int u32x4 __attribute__((ext_vector_type(4)));
typedef int          i32x8 __attribute__((ext_vector_type(8)));
typedef int          i32x4 __attribute__((ext_vector_type(4)));

union Frag { v16bf v; uint4 q[2]; };

__device__ __forceinline__ unsigned short f32_to_bf16_rne(float f) {
    unsigned int u = __float_as_uint(f);
    u += 0x7FFFu + ((u >> 16) & 1u);   // round-to-nearest-even
    return (unsigned short)(u >> 16);
}

// --------------------------- kernel 0: zero out + counts --------------------
__global__ void moe_zero_kernel(float* __restrict__ out, int* __restrict__ counts) {
    const int total = MOE_N * MOE_D;
    for (int i = blockIdx.x * blockDim.x + threadIdx.x; i < total;
         i += gridDim.x * blockDim.x) {
        out[i] = 0.0f;
    }
    if (blockIdx.x == 0 && threadIdx.x < MOE_E) counts[threadIdx.x] = 0;
}

// --------------------------- kernel 1: x -> bf16 (vectorized) ---------------
__global__ void moe_cvt_x_kernel(const float* __restrict__ x,
                                 unsigned short* __restrict__ xb) {
    const int total4 = MOE_N * MOE_D / 4;
    for (int i = blockIdx.x * blockDim.x + threadIdx.x; i < total4;
         i += gridDim.x * blockDim.x) {
        const float4 v = ((const float4*)x)[i];
        uint2 o;
        o.x = (unsigned)f32_to_bf16_rne(v.x) | ((unsigned)f32_to_bf16_rne(v.y) << 16);
        o.y = (unsigned)f32_to_bf16_rne(v.z) | ((unsigned)f32_to_bf16_rne(v.w) << 16);
        ((uint2*)xb)[i] = o;
    }
}

// ------------------- kernel 2: W[e][d][h] -> Wt[e][h][d] bf16 ---------------
__global__ void moe_cvt_w_kernel(const float* __restrict__ W,
                                 unsigned short* __restrict__ Wt) {
    __shared__ unsigned short T[64][65];
    const int e  = blockIdx.z;
    const int dt = blockIdx.y;
    const int ht = blockIdx.x;
    const int t  = threadIdx.x;
    const int c4 = (t & 15) * 4;
    const int r0 = t >> 4;

#pragma unroll
    for (int pass = 0; pass < 4; ++pass) {
        const int i = r0 + pass * 16;   // d-local
        const float4 w = *(const float4*)(
            W + (((size_t)e * MOE_D) + dt * 64 + i) * MOE_D + ht * 64 + c4);
        T[i][c4 + 0] = f32_to_bf16_rne(w.x);
        T[i][c4 + 1] = f32_to_bf16_rne(w.y);
        T[i][c4 + 2] = f32_to_bf16_rne(w.z);
        T[i][c4 + 3] = f32_to_bf16_rne(w.w);
    }
    __syncthreads();
#pragma unroll
    for (int pass = 0; pass < 4; ++pass) {
        const int j = r0 + pass * 16;   // h-local
        uint2 o;
        o.x = (unsigned)T[c4 + 0][j] | ((unsigned)T[c4 + 1][j] << 16);
        o.y = (unsigned)T[c4 + 2][j] | ((unsigned)T[c4 + 3][j] << 16);
        *(uint2*)(Wt + (((size_t)e * MOE_D) + ht * 64 + j) * MOE_D + dt * 64 + c4) = o;
    }
}

// --------------------------- kernel 3: gating + top-2 scatter ---------------
__global__ void moe_gate_kernel(const float* __restrict__ x,
                                const float* __restrict__ Wg,   // [D][E]
                                const float* __restrict__ bg,   // [E]
                                int*   __restrict__ counts,
                                int*   __restrict__ btok,       // [E][N]
                                float* __restrict__ bgate) {    // [E][N]
    const int lane = threadIdx.x & 31;
    const int wave = threadIdx.x >> 5;
    const int n = blockIdx.x * 8 + wave;
    if (n >= MOE_N) return;

    float acc[MOE_E];
#pragma unroll
    for (int e = 0; e < MOE_E; ++e) acc[e] = 0.0f;

    const float* xr = x + (size_t)n * MOE_D;
    for (int d = lane; d < MOE_D; d += 32) {
        const float xv = xr[d];
        const float* wr = Wg + (size_t)d * MOE_E;
#pragma unroll
        for (int e = 0; e < MOE_E; ++e) acc[e] += xv * wr[e];
    }
#pragma unroll
    for (int off = 16; off >= 1; off >>= 1) {
#pragma unroll
        for (int e = 0; e < MOE_E; ++e)
            acc[e] += __shfl_xor(acc[e], off, 32);
    }

    if (lane == 0) {
        float lg[MOE_E];
        float m = -3.0e38f;
#pragma unroll
        for (int e = 0; e < MOE_E; ++e) { lg[e] = acc[e] + bg[e]; m = fmaxf(m, lg[e]); }
        float p[MOE_E];
        float s = 0.0f;
#pragma unroll
        for (int e = 0; e < MOE_E; ++e) { p[e] = expf(lg[e] - m); s += p[e]; }
        const float inv = 1.0f / s;

        int i0 = 0; float v0 = p[0];
#pragma unroll
        for (int e = 1; e < MOE_E; ++e) if (p[e] > v0) { v0 = p[e]; i0 = e; }
        int i1 = -1; float v1 = -1.0f;
#pragma unroll
        for (int e = 0; e < MOE_E; ++e)
            if (e != i0 && p[e] > v1) { v1 = p[e]; i1 = e; }

        int s0 = atomicAdd(&counts[i0], 1);
        btok[i0 * MOE_N + s0] = n;  bgate[i0 * MOE_N + s0] = v0 * inv;
        int s1 = atomicAdd(&counts[i1], 1);
        btok[i1 * MOE_N + s1] = n;  bgate[i1 * MOE_N + s1] = v1 * inv;
    }
}

// --------------------------- kernel 4: grouped WMMA GEMM --------------------
// Block tile 128(M) x 64(N); 8 waves = 4(M) x 2(N); wave tile 32x32 ->
// 4 x v_wmma per K-step over D/32 steps. The 64x512 bf16 B panel is loaded
// ONCE per block by the Tensor Data Mover into LDS (row stride padded to
// 1040B via TDM pad so fragment loads are bank-conflict-free); only the
// gathered A rows are software-pipelined through LDS per K-step.
#define B_ROW_BYTES 1040   // 512*2 + 16 pad

__global__ __launch_bounds__(256)
void moe_gemm_kernel(const unsigned short* __restrict__ xb,   // bf16 [N][D]
                     const unsigned short* __restrict__ Wt,   // bf16 [E][h][d]
                     const float* __restrict__ b,             // [E][D]
                     const int*   __restrict__ counts,
                     const int*   __restrict__ btok,
                     const float* __restrict__ bgate,
                     float* __restrict__ out) {               // [N][D]
    const int e  = blockIdx.z;
    const int ct = blockIdx.x;          // column tile: cols ct*64 .. +63
    const int rt = blockIdx.y;          // row tile within expert bucket
    const int cnt = counts[e];
    const int row_base = rt * 128;
    if (row_base >= cnt) return;

    __shared__ unsigned short As[128][40];            // [m][k], 80B rows
    __shared__ unsigned char  Bp[64 * B_ROW_BYTES];   // TDM-padded B panel
    __shared__ int   tokS[128];
    __shared__ float gateS[128];

    const int t    = threadIdx.x;
    const int lane = t & 31;
    const int wave = t >> 5;
    const int wm   = wave >> 1;         // 0..3 -> rows 32*wm
    const int wn   = wave & 1;          // 0..1 -> cols 32*wn
    const int hh   = lane >> 4;         // K-half select
    const int ln   = lane & 15;

    // ---- TDM: DMA the whole B panel (64 rows x 512 k, bf16) into LDS ------
    if (t < 32) {                       // one wave issues (TDM ignores EXEC)
        const unsigned long long gaddr =
            (unsigned long long)(const void*)(Wt + (((size_t)e * MOE_D) + ct * 64) * MOE_D);
        const unsigned lds_addr = (unsigned)(uintptr_t)&Bp[0];  // low 32b = LDS offset

        u32x4 g0;
        g0.x = 1u;                                    // count=1, user descriptor
        g0.y = lds_addr;                              // LDS dest (bytes)
        g0.z = (unsigned)(gaddr & 0xFFFFFFFFu);       // global_addr[31:0]
        g0.w = (unsigned)((gaddr >> 32) & 0x01FFFFFFu) | (2u << 30);  // [56:32]|type=2

        i32x8 g1;
        g1[0] = (1 << 16)      // data_size = 2 bytes
              | (1 << 20)      // pad_enable
              | (7 << 22)      // pad_interval: 256 DWORDs (=1024B, one row)
              | (3 << 25);     // pad_amount: 4 DWORDs (=16B) -> 1040B row stride
        g1[1] = (MOE_D & 0xFFFF) << 16;               // tensor_dim0[15:0] = 512
        g1[2] = (MOE_D >> 16) | (64 << 16);           // dim0[31:16] | tensor_dim1=64
        g1[3] = (MOE_D & 0xFFFF) << 16;               // dim1[31:16]=0 | tile_dim0=512
        g1[4] = 64;                                   // tile_dim1=64 | tile_dim2=0
        g1[5] = MOE_D;                                // tensor_dim0_stride = 512
        g1[6] = 0;
        g1[7] = 0;

        i32x4 gz;   gz[0] = 0; gz[1] = 0; gz[2] = 0; gz[3] = 0;
        i32x8 gz8;  gz8[0] = 0; gz8[1] = 0; gz8[2] = 0; gz8[3] = 0;
                    gz8[4] = 0; gz8[5] = 0; gz8[6] = 0; gz8[7] = 0;

        // clang-23 / amdgpu-toolchain 6-arg form:
        // (g0 u32x4, g1 i32x8, g2 i32x4, g3 i32x4, i32x8, cpol)
        __builtin_amdgcn_tensor_load_to_lds(g0, g1, gz, gz, gz8, 0);
        __builtin_amdgcn_s_wait_tensorcnt(0);
    }

    if (t < 128) {
        const int idx = row_base + t;
        const bool ok = idx < cnt;
        tokS[t]  = ok ? btok [e * MOE_N + idx] : 0;
        gateS[t] = ok ? bgate[e * MOE_N + idx] : 0.0f;
    }
    __syncthreads();                    // B panel + token list visible to all

    // ---- A staging assignment (per-thread, constant across K-loop) --------
    const int ar = t >> 1;              // A row 0..127
    const int ak = (t & 1) * 16;        // A k-offset {0,16}, 16 bf16
    const unsigned short* aglob = xb + (size_t)tokS[ar] * MOE_D + ak;

    const v8f zf = {0.f, 0.f, 0.f, 0.f, 0.f, 0.f, 0.f, 0.f};
    v8f acc[2][2] = {{zf, zf}, {zf, zf}};

    uint4 a0 = *(const uint4*)(aglob);
    uint4 a1 = *(const uint4*)(aglob + 8);

    // per-wave B row bases in the padded panel
    const unsigned char* brow0 = Bp + (size_t)(32 * wn + ln) * B_ROW_BYTES;       // ns=0
    const unsigned char* brow1 = Bp + (size_t)(32 * wn + 16 + ln) * B_ROW_BYTES;  // ns=1

    for (int kt = 0; kt < MOE_D / 32; ++kt) {
        __syncthreads();                 // previous A tile fully consumed
        *(uint4*)&As[ar][ak]     = a0;
        *(uint4*)&As[ar][ak + 8] = a1;
        if (kt + 1 < MOE_D / 32) {       // prefetch next A tile (overlaps WMMA)
            const int k0n = (kt + 1) * 32;
            a0 = *(const uint4*)(aglob + k0n);
            a1 = *(const uint4*)(aglob + k0n + 8);
        }
        __syncthreads();

        const int kbyte = kt * 64;       // k0*2 bytes into each B row
        Frag af[2], bf[2];
#pragma unroll
        for (int ms = 0; ms < 2; ++ms) {
            const int mA = 32 * wm + 16 * ms + ln;
            af[ms].q[0] = *(const uint4*)&As[mA][16 * hh];      // K 16h..16h+7
            af[ms].q[1] = *(const uint4*)&As[mA][16 * hh + 8];  // K 16h+8..+15
        }
        bf[0].q[0] = *(const uint4*)(brow0 + kbyte + 32 * hh);      // K 16h..16h+7
        bf[0].q[1] = *(const uint4*)(brow0 + kbyte + 32 * hh + 16); // K 16h+8..+15
        bf[1].q[0] = *(const uint4*)(brow1 + kbyte + 32 * hh);
        bf[1].q[1] = *(const uint4*)(brow1 + kbyte + 32 * hh + 16);

#pragma unroll
        for (int ms = 0; ms < 2; ++ms)
#pragma unroll
            for (int ns = 0; ns < 2; ++ns)
                acc[ms][ns] = __builtin_amdgcn_wmma_f32_16x16x32_bf16(
                    false, af[ms].v, false, bf[ns].v, (short)0, acc[ms][ns],
                    false, false);
    }

    // ---- epilogue: out[token] += g * (acc + b[e]) ----
    const int col0 = ct * 64 + 32 * wn + ln;
    const float bias0 = b[e * MOE_D + col0];
    const float bias1 = b[e * MOE_D + col0 + 16];
#pragma unroll
    for (int ms = 0; ms < 2; ++ms) {
#pragma unroll
        for (int r = 0; r < 8; ++r) {
            const int lr = 32 * wm + 16 * ms + hh * 8 + r;
            if (row_base + lr < cnt) {
                const int   token = tokS[lr];
                const float g     = gateS[lr];
                float* orow = out + (size_t)token * MOE_D;
                atomicAdd(orow + col0,      g * (acc[ms][0][r] + bias0));
                atomicAdd(orow + col0 + 16, g * (acc[ms][1][r] + bias1));
            }
        }
    }
}

// ---------------------------------------------------------------------------
extern "C" void kernel_launch(void* const* d_in, const int* in_sizes, int n_in,
                              void* d_out, int out_size, void* d_ws, size_t ws_size,
                              hipStream_t stream) {
    const float* x  = (const float*)d_in[0];   // [N,D]
    const float* W  = (const float*)d_in[1];   // [E,D,D]
    const float* b  = (const float*)d_in[2];   // [E,D]
    const float* Wg = (const float*)d_in[3];   // [D,E]
    const float* bg = (const float*)d_in[4];   // [E]
    float* out = (float*)d_out;

    // workspace layout (~40 MB)
    char* ws = (char*)d_ws;
    const size_t XB_OFF   = 0;                                    // N*D bf16
    const size_t WT_OFF   = XB_OFF + (size_t)MOE_N * MOE_D * 2;   // E*D*D bf16
    const size_t CNT_OFF  = WT_OFF + (size_t)MOE_E * MOE_D * MOE_D * 2;
    const size_t TOK_OFF  = CNT_OFF + 1024;                       // E*N int
    const size_t GATE_OFF = TOK_OFF + (size_t)MOE_E * MOE_N * 4;  // E*N float

    unsigned short* xb     = (unsigned short*)(ws + XB_OFF);
    unsigned short* Wt     = (unsigned short*)(ws + WT_OFF);
    int*            counts = (int*)(ws + CNT_OFF);
    int*            btok   = (int*)(ws + TOK_OFF);
    float*          bgate  = (float*)(ws + GATE_OFF);

    moe_zero_kernel<<<8192, 256, 0, stream>>>(out, counts);
    moe_cvt_x_kernel<<<4096, 256, 0, stream>>>(x, xb);
    {
        dim3 g(MOE_D / 64, MOE_D / 64, MOE_E);    // 8 x 8 x 8
        moe_cvt_w_kernel<<<g, 256, 0, stream>>>(W, Wt);
    }
    moe_gate_kernel<<<MOE_N / 8, 256, 0, stream>>>(x, Wg, bg, counts, btok, bgate);
    {
        dim3 g(MOE_D / 64, MOE_N / 128, MOE_E);   // 8 x 256 x 8; empty tiles exit
        moe_gemm_kernel<<<g, 256, 0, stream>>>(xb, Wt, b, counts, btok, bgate, out);
    }
}
